// Int8Linear_91199335563392
// MI455X (gfx1250) — compile-verified, matrix operand order checked by reading, not run on hardware
//
#include <hip/hip_runtime.h>
#include <stdint.h>

typedef __attribute__((ext_vector_type(16))) _Float16 v16h;
typedef __attribute__((ext_vector_type(8)))  _Float16 v8h;
typedef __attribute__((ext_vector_type(2)))  _Float16 h2;
typedef __attribute__((ext_vector_type(8)))  float    v8f;
typedef __attribute__((ext_vector_type(8)))  unsigned int v8u;

#define IN_F    4096
#define OUT_F   16384
#define KCHUNK  1024         // K staged in LDS per outer iteration (32 KB f16)
#define WAVES   4            // waves per block, 1 N-tile (16 cols) per wave
#define TPB     (WAVES * 32)
#define GRP_SH  7            // log2(KCHUNK/8) groups-of-8 per row
#define GRP_MSK ((KCHUNK / 8) - 1)

// Dequant 2 biased uint8 bytes (selected by `sel` out of xb) to packed f16:
//   v_perm_b32   -> {0x64|b_lo, 0x64|b_hi}  (f16 bits of 1024+b, exact)
//   v_pk_add_f16 -1152 -> exact int8 value as f16 pair
static __device__ __forceinline__ unsigned int dq2(unsigned int xb, unsigned int sel) {
    unsigned int r = __builtin_amdgcn_perm(0x64646464u, xb, sel);
    h2 h = __builtin_bit_cast(h2, r);
    const h2 off = { (_Float16)(-1152.0f), (_Float16)(-1152.0f) };
    h = h + off;                      // v_pk_add_f16
    return __builtin_bit_cast(unsigned int, h);
}

__global__ __launch_bounds__(TPB)
void int8linear_wmma(const float* __restrict__ x,
                     const int8_t* __restrict__ w,
                     const float* __restrict__ scale,
                     const float* __restrict__ bias,
                     float* __restrict__ out)
{
    // x[:, kc:kc+KCHUNK] staged as f16, row-major, row stride KCHUNK
    __shared__ _Float16 xs[16 * KCHUNK];   // 32 KB

    const int t    = threadIdx.x;
    const int lane = t & 31;
    const int wave = t >> 5;

    const int nbase = (blockIdx.x * WAVES + wave) * 16; // 16 output cols per wave
    const int mrow  = lane & 15;   // A: row M this lane reads; B/C: column index
    const int hi    = lane >> 4;   // lane-half selects K-run offset (+8 / +24)
    const int ncol  = nbase + mrow;

    const int8_t* wrow = w + (size_t)ncol * IN_F;   // contiguous int8 W row per lane

    v8f acc = {};

    for (int kc = 0; kc < IN_F; kc += KCHUNK) {
        // ---- cooperative stage: x f32 -> f16 into LDS ----
        __syncthreads();                       // protect previous chunk's readers
        for (int g = t; g < (16 * KCHUNK) / 8; g += TPB) {
            const int row = g >> GRP_SH;       // KCHUNK/8 groups of 8 per row
            const int k8  = (g & GRP_MSK) * 8;
            const float4 f0 = *(const float4*)(x + (size_t)row * IN_F + kc + k8);
            const float4 f1 = *(const float4*)(x + (size_t)row * IN_F + kc + k8 + 4);
            v8h h;
            h[0] = (_Float16)f0.x; h[1] = (_Float16)f0.y;
            h[2] = (_Float16)f0.z; h[3] = (_Float16)f0.w;
            h[4] = (_Float16)f1.x; h[5] = (_Float16)f1.y;
            h[6] = (_Float16)f1.z; h[7] = (_Float16)f1.w;
            *(v8h*)(&xs[row * KCHUNK + k8]) = h;   // ds_store_b128
        }
        __syncthreads();

        // prefetch next chunk of this lane's weight row into the caches
        if (kc + KCHUNK < IN_F)
            __builtin_prefetch(wrow + kc + KCHUNK + hi * 64, 0, 0);

        // ---- KCHUNK/32 WMMA steps over this K chunk ----
        #pragma unroll 4
        for (int ks = 0; ks < KCHUNK / 32; ++ks) {
            const int kb = ks * 32 + hi * 8;   // local K base of this lane's runs

            // A fragment: two contiguous 8xf16 K-runs from LDS (ds_load_b128 x2)
            const v8h a0 = *(const v8h*)(&xs[mrow * KCHUNK + kb]);
            const v8h a1 = *(const v8h*)(&xs[mrow * KCHUNK + kb + 16]);
            const v16h a = __builtin_shufflevector(a0, a1,
                0,1,2,3,4,5,6,7,8,9,10,11,12,13,14,15);

            // B fragment: two 8-byte int8 runs from the weight stream
            const uint2 r0 = *(const uint2*)(wrow + kc + kb);
            const uint2 r1 = *(const uint2*)(wrow + kc + kb + 16);

            // bias int8 -> uint8 (x + 128) with one xor per dword
            const unsigned int x0 = r0.x ^ 0x80808080u;
            const unsigned int x1 = r0.y ^ 0x80808080u;
            const unsigned int x2 = r1.x ^ 0x80808080u;
            const unsigned int x3 = r1.y ^ 0x80808080u;

            v8u pb;
            pb[0] = dq2(x0, 0x05010400u);      // bytes {0,1}
            pb[1] = dq2(x0, 0x05030402u);      // bytes {2,3}
            pb[2] = dq2(x1, 0x05010400u);
            pb[3] = dq2(x1, 0x05030402u);
            pb[4] = dq2(x2, 0x05010400u);
            pb[5] = dq2(x2, 0x05030402u);
            pb[6] = dq2(x3, 0x05010400u);
            pb[7] = dq2(x3, 0x05030402u);
            const v16h b = __builtin_bit_cast(v16h, pb);

            // D = A x B + C   (f32 accumulate)
            acc = __builtin_amdgcn_wmma_f32_16x16x32_f16(
                      /*neg_a=*/false, a, /*neg_b=*/false, b,
                      /*c_mod=*/(short)0, acc,
                      /*reuse_a=*/false, /*reuse_b=*/false);
        }
    }

    // ---- epilogue: y = acc * scale + bias, per C/D VGPR layout ----
    const float s  = *scale;
    const float bb = bias[ncol];
    #pragma unroll
    for (int r = 0; r < 8; ++r) {
        const int m = r + hi * 8;              // lanes 16-31 hold rows M=8..15
        out[(size_t)m * OUT_F + ncol] = acc[r] * s + bb;
    }
}

extern "C" void kernel_launch(void* const* d_in, const int* in_sizes, int n_in,
                              void* d_out, int out_size, void* d_ws, size_t ws_size,
                              hipStream_t stream) {
    const float*  x  = (const float*)d_in[0];   // [16, 4096] f32
    const int8_t* w  = (const int8_t*)d_in[1];  // [16384, 4096] int8
    const float*  s  = (const float*)d_in[2];   // scalar scale
    const float*  b  = (const float*)d_in[3];   // [16384] f32
    float* out = (float*)d_out;                 // [16, 16384] f32

    dim3 grid(OUT_F / (WAVES * 16));            // 256 blocks x 4 waves = 1024 tiles
    dim3 block(TPB);
    hipLaunchKernelGGL(int8linear_wmma, grid, block, 0, stream, x, w, s, b, out);
}
